// VoiceDDSP_Training_63565515981200
// MI455X (gfx1250) — matrix operand order repaired and checked by reading, not compiled, not run on hardware
//
#include <hip/hip_runtime.h>
#include <hip/hip_bf16.h>
#include <math.h>

// ---------------- constants ----------------
#define SRf   24000.0f
#define HOP   256
#define KTAP  1024
#define RCH   512
#define FRM   2048          // frames per batch
#define NB    4             // batch
#define LCONV 1279          // HOP + KTAP - 1
#define TL    525311        // (FRM-1)*HOP + LCONV
#define TLPAD 525312
#define REVN  24000
#define PI_F  3.14159265358979323846f

typedef __attribute__((ext_vector_type(16))) _Float16 v16h;
typedef __attribute__((ext_vector_type(8)))  float    v8f;
typedef __attribute__((ext_vector_type(4)))  int      v4i;
typedef _Float16 h16;

union AF { v16h v; h16 h[16]; };
union CF { v8f  v; float f[8]; };

// -------- CDNA5 async global->LDS path (guarded; falls back if absent) --------
#if defined(__has_builtin)
#if __has_builtin(__builtin_amdgcn_global_load_async_to_lds_b128) && \
    __has_builtin(__builtin_amdgcn_s_wait_asynccnt)
#define USE_ASYNC_LDS 1
#endif
#endif

#ifdef USE_ASYNC_LDS
#define AS1 __attribute__((address_space(1)))
#define AS3 __attribute__((address_space(3)))
__device__ inline void lds_async_b128(void* lds_dst, const void* gsrc) {
  __builtin_amdgcn_global_load_async_to_lds_b128(
      (AS1 v4i*)(uintptr_t)gsrc,
      (AS3 v4i*)(unsigned int)(uintptr_t)lds_dst,
      0, 0);
}
__device__ inline void wait_async0() { __builtin_amdgcn_s_wait_asynccnt(0); }
#endif

// ---------------- WMMA fragment loaders (CDNA5 16-bit layouts) ----------------
// A (16x32 f16): lane L: M=L&15, half=L>>4; VGPR0-7 hold K=half*16+0..15
// (one contiguous 16-half chunk, matching packed B).
__device__ inline v16h load_a_frag(const h16* A, int lda, int m0, int k0, int lane) {
  AF a;
  int M = m0 + (lane & 15);
  int hs = lane >> 4;
  const h16* p = A + (size_t)M * lda + k0 + hs * 16;
#pragma unroll
  for (int i = 0; i < 8; i++) a.h[i] = p[i];
#pragma unroll
  for (int i = 0; i < 8; i++) a.h[8 + i] = p[8 + i];
  return a.v;
}
// B (32x16 f16) pre-packed so lane L reads 16 contiguous halves (32B):
// packed[lane*16 + i] = B[K = (L>>4)*16 + i][N = L&15]
__device__ inline v16h load_b_frag(const h16* Bp, int lane) {
  AF b;
  const h16* p = Bp + lane * 16;
#pragma unroll
  for (int i = 0; i < 16; i++) b.h[i] = p[i];
  return b.v;
}

__device__ inline float softplusf(float x) { return x > 20.f ? x : log1pf(__expf(x)); }
__device__ inline float sigmoidf_(float x) { return 1.f / (1.f + __expf(-x)); }

// ---------------- small prep kernels ----------------
__global__ void mean_f0_kernel(const float* __restrict__ f0, float* __restrict__ mean) {
  __shared__ float red[256];
  int b = blockIdx.x, tid = threadIdx.x;
  float s = 0.f;
  for (int i = tid; i < FRM; i += 256) s += f0[b * FRM + i];
  red[tid] = s; __syncthreads();
  for (int st = 128; st > 0; st >>= 1) { if (tid < st) red[tid] += red[tid + st]; __syncthreads(); }
  if (tid == 0) mean[b] = red[0] / (float)FRM;
}

// inclusive scan of cycles per batch; start = frac(cum - cycles)
__global__ void __launch_bounds__(1024) scan_start_kernel(const float* __restrict__ f0,
                                                          float* __restrict__ start) {
  __shared__ float s[FRM];
  int b = blockIdx.x, tid = threadIdx.x;
  int i0 = tid, i1 = tid + 1024;
  float c0 = f0[b * FRM + i0] * ((float)HOP / SRf);
  float c1 = f0[b * FRM + i1] * ((float)HOP / SRf);
  s[i0] = c0; s[i1] = c1; __syncthreads();
  for (int off = 1; off < FRM; off <<= 1) {
    float v0 = (i0 >= off) ? s[i0 - off] : 0.f;
    float v1 = (i1 >= off) ? s[i1 - off] : 0.f;
    __syncthreads();
    s[i0] += v0; s[i1] += v1;
    __syncthreads();
  }
  float e0 = s[i0] - c0, e1 = s[i1] - c1;
  start[b * FRM + i0] = e0 - floorf(e0);
  start[b * FRM + i1] = e1 - floorf(e1);
}

__global__ void build_x28_kernel(const float* __restrict__ f0, const float* __restrict__ loud,
                                 const float* __restrict__ gender, const float* __restrict__ age,
                                 const float* __restrict__ z, const float* __restrict__ mean,
                                 h16* __restrict__ x28) {
  int row = blockIdx.x * blockDim.x + threadIdx.x;
  if (row >= NB * FRM) return;
  int b = row / FRM;
  h16* o = x28 + (size_t)row * 32;
  o[0] = (h16)(f0[row] - mean[b]);
  o[1] = (h16)loud[row];
  o[2] = (h16)gender[row * 2 + 0];
  o[3] = (h16)gender[row * 2 + 1];
#pragma unroll
  for (int i = 0; i < 8; i++)  o[4 + i]  = (h16)age[row * 8 + i];
#pragma unroll
  for (int i = 0; i < 16; i++) o[12 + i] = (h16)z[row * 16 + i];
#pragma unroll
  for (int i = 28; i < 32; i++) o[i] = (h16)0.f;
}

// Pack W[N][K] (row-major, ldw, valid K = kvalid) into WMMA-B fragment order.
__global__ void pack_b_kernel(const float* __restrict__ W, int ldw, int kvalid,
                              int Ntiles, int KT, h16* __restrict__ Bp) {
  int idx = blockIdx.x * blockDim.x + threadIdx.x;
  int total = Ntiles * KT * 32;
  if (idx >= total) return;
  int nt = idx / (KT * 32);
  int rem = idx - nt * (KT * 32);
  int kt = rem >> 5;
  int lane = rem & 31;
  int N = nt * 16 + (lane & 15);
  int kbase = kt * 32 + (lane >> 4) * 16;
  h16* out = Bp + ((size_t)(nt * KT + kt) * 32 + lane) * 16;
#pragma unroll
  for (int i = 0; i < 16; i++) {
    int k = kbase + i;
    out[i] = (h16)((k < kvalid) ? W[(size_t)N * ldw + k] : 0.f);
  }
}

// ---------------- generic WMMA GEMM: C[M][N] = A[M][K] * B + bias ----------------
__global__ void wmma_gemm_kernel(const h16* __restrict__ A, int lda,
                                 const h16* __restrict__ Bp, int KT,
                                 const float* __restrict__ bias,
                                 float* __restrict__ Cf, h16* __restrict__ Ch,
                                 int ldc, int Ntiles, int relu) {
  int gw = (blockIdx.x * blockDim.x + threadIdx.x) >> 5;
  int lane = threadIdx.x & 31;
  int mt = gw / Ntiles, nt = gw % Ntiles;
  v8f acc = {0.f, 0.f, 0.f, 0.f, 0.f, 0.f, 0.f, 0.f};
  const h16* bp = Bp + (size_t)nt * KT * 512;
  for (int kt = 0; kt < KT; kt++) {
    v16h a = load_a_frag(A, lda, mt * 16, kt * 32, lane);
    v16h b = load_b_frag(bp + (size_t)kt * 512, lane);
    acc = __builtin_amdgcn_wmma_f32_16x16x32_f16(false, a, false, b, (short)0, acc, false, false);
  }
  CF c; c.v = acc;
  int n = nt * 16 + (lane & 15);
  float bv = bias ? bias[n] : 0.f;
#pragma unroll
  for (int r = 0; r < 8; r++) {
    int m = mt * 16 + r + 8 * (lane >> 4);
    float v = c.f[r] + bv;
    if (relu) v = fmaxf(v, 0.f);
    if (Cf) Cf[(size_t)m * ldc + n] = v;
    if (Ch) Ch[(size_t)m * ldc + n] = (h16)v;
  }
}

// ---------------- persistent single-WGP GRU (WMMA recurrence) ----------------
// Per step: async-prefetch GI[t] (24KB) into LDS while 96x16 WMMAs compute
// gh = h * W_hh^T; then fused gates in LDS. Only s_barrier between phases.
__global__ void __launch_bounds__(1024) gru_kernel(const h16* __restrict__ Bp_hh,
                                                   const float* __restrict__ GI,
                                                   const float* __restrict__ b_hh,
                                                   float* __restrict__ Hall) {
  __shared__ h16  hsh[16 * RCH];      // A matrix: rows 0..3 = batch h (f16), 4..15 zero
  __shared__ float ghs[NB * 3 * RCH]; // gh for this step
  __shared__ float hf[NB * RCH];      // fp32 hidden state
#ifdef USE_ASYNC_LDS
  __shared__ float gis[NB * 3 * RCH]; // async-staged GI rows for this step
#endif
  int tid = threadIdx.x, lane = tid & 31, wave = tid >> 5; // 32 waves
  for (int i = tid; i < 16 * RCH; i += 1024) hsh[i] = (h16)0.f;
  for (int i = tid; i < NB * RCH; i += 1024) hf[i] = 0.f;
  __syncthreads();
  const int KT = RCH / 32; // 16
  for (int t = 0; t < FRM; t++) {
#ifdef USE_ASYNC_LDS
    // 1536 x 16B async transfers: GI rows (b*FRM+t) -> LDS, hidden under WMMAs
    for (int i = tid; i < NB * 3 * RCH / 4; i += 1024) {
      int b = i / 384;               // 1536 floats per row / 4 per b128
      int off = (i - b * 384) * 4;
      lds_async_b128(gis + b * (3 * RCH) + off,
                     GI + (size_t)(b * FRM + t) * (3 * RCH) + off);
    }
#endif
    // each wave computes 3 of the 96 N-tiles (1536 columns)
#pragma unroll
    for (int q = 0; q < 3; q++) {
      int nt = wave * 3 + q;
      v8f acc = {0.f, 0.f, 0.f, 0.f, 0.f, 0.f, 0.f, 0.f};
      const h16* bp = Bp_hh + (size_t)nt * KT * 512;
      for (int kt = 0; kt < KT; kt++) {
        v16h a = load_a_frag(hsh, RCH, 0, kt * 32, lane);
        v16h b = load_b_frag(bp + (size_t)kt * 512, lane);
        acc = __builtin_amdgcn_wmma_f32_16x16x32_f16(false, a, false, b, (short)0, acc, false, false);
      }
      CF c; c.v = acc;
      int n = nt * 16 + (lane & 15);
      int mb = 8 * (lane >> 4);
#pragma unroll
      for (int r = 0; r < 8; r++) {
        int m = mb + r;
        if (m < NB) ghs[m * (3 * RCH) + n] = c.f[r];
      }
    }
#ifdef USE_ASYNC_LDS
    wait_async0();
#endif
    __syncthreads();
    if (tid < RCH) {
      int j = tid;
#pragma unroll
      for (int b = 0; b < NB; b++) {
#ifdef USE_ASYNC_LDS
        const float* gi = gis + b * (3 * RCH);
#else
        const float* gi = GI + (size_t)(b * FRM + t) * (3 * RCH);
#endif
        const float* gh = ghs + b * (3 * RCH);
        float r = sigmoidf_(gi[j] + gh[j] + b_hh[j]);
        float zg = sigmoidf_(gi[RCH + j] + gh[RCH + j] + b_hh[RCH + j]);
        float n = tanhf(gi[2 * RCH + j] + r * (gh[2 * RCH + j] + b_hh[2 * RCH + j]));
        float hp = hf[b * RCH + j];
        float hn = (1.f - zg) * n + zg * hp;
        hf[b * RCH + j] = hn;
        hsh[b * RCH + j] = (h16)hn;
        Hall[(size_t)(b * FRM + t) * RCH + j] = hn;
      }
    }
    __syncthreads();
  }
}

// ---------------- 11 output heads ----------------
__global__ void heads_kernel(const float* __restrict__ H,
                             const float* Woq, const float* boq,
                             const float* Wvg, const float* bvg,
                             const float* Wug, const float* bug,
                             const float* Wff, const float* bff,
                             const float* Wfb, const float* bfb,
                             float* __restrict__ oq, float* __restrict__ vg,
                             float* __restrict__ ug, float* __restrict__ ff,
                             float* __restrict__ fb) {
  __shared__ float hrow[RCH];
  __shared__ float red[11 * 256];
  int row = blockIdx.x, tid = threadIdx.x;
  const float* hp = H + (size_t)row * RCH;
  hrow[tid] = hp[tid]; hrow[tid + 256] = hp[tid + 256];
  __syncthreads();
  const float* w[11] = { Woq, Wvg, Wug,
                         Wff, Wff + 512, Wff + 1024, Wff + 1536,
                         Wfb, Wfb + 512, Wfb + 1024, Wfb + 1536 };
#pragma unroll
  for (int h = 0; h < 11; h++) {
    float p = hrow[tid] * w[h][tid] + hrow[tid + 256] * w[h][tid + 256];
    red[h * 256 + tid] = p;
  }
  __syncthreads();
  for (int s = 128; s > 0; s >>= 1) {
    if (tid < s)
#pragma unroll
      for (int h = 0; h < 11; h++) red[h * 256 + tid] += red[h * 256 + tid + s];
    __syncthreads();
  }
  if (tid == 0) {
    oq[row] = sigmoidf_(red[0 * 256] + boq[0]);
    vg[row] = softplusf(red[1 * 256] + bvg[0]);
    ug[row] = softplusf(red[2 * 256] + bug[0]);
    float c = 200.f;
#pragma unroll
    for (int i = 0; i < 4; i++) {
      c += softplusf(red[(3 + i) * 256] + bff[i]);
      ff[row * 4 + i] = c;
      fb[row * 4 + i] = softplusf(red[(7 + i) * 256] + bfb[i]) + 50.f;
    }
  }
}

// ---------------- Rosenberg glottal source + excitation ----------------
__global__ void excitation_kernel(const float* __restrict__ f0, const float* __restrict__ start,
                                  const float* __restrict__ noise,
                                  const float* __restrict__ oq, const float* __restrict__ vg,
                                  const float* __restrict__ ug, float* __restrict__ exc) {
  int row = blockIdx.x, s = threadIdx.x;
  float f0v = f0[row];
  float tot = start[row] + f0v * ((float)s / SRf);
  float phi = tot - floorf(tot);
  float oqv = oq[row];
  float peak = 0.66f * oqv;
  const float eps = 1e-6f;
  float rise = 0.5f * (1.f - __cosf(PI_F * phi / (peak + eps)));
  float fall = __cosf(PI_F * (phi - peak) / (2.f * (oqv - peak) + eps));
  float voiced = (phi < peak) ? rise : ((phi < oqv) ? fall : 0.f);
  exc[(size_t)row * HOP + s] = voiced * vg[row] + noise[(size_t)row * HOP + s] * ug[row];
}

// ---------------- time-varying FIR build (normalized, pre-reversed) ----------------
__global__ void filter_kernel(const float* __restrict__ ff, const float* __restrict__ fb,
                              float* __restrict__ filtrev) {
  __shared__ float comb[KTAP];
  __shared__ float red[256];
  int row = blockIdx.x, tid = threadIdx.x;
  float f0_ = ff[row * 4 + 0], f1_ = ff[row * 4 + 1], f2_ = ff[row * 4 + 2], f3_ = ff[row * 4 + 3];
  float b0_ = fb[row * 4 + 0], b1_ = fb[row * 4 + 1], b2_ = fb[row * 4 + 2], b3_ = fb[row * 4 + 3];
  float mx = 0.f;
  for (int k = tid; k < KTAP; k += 256) {
    float t = (float)k / SRf;
    float v = __expf(-PI_F * b0_ * t) * __sinf(2.f * PI_F * f0_ * t)
            + __expf(-PI_F * b1_ * t) * __sinf(2.f * PI_F * f1_ * t)
            + __expf(-PI_F * b2_ * t) * __sinf(2.f * PI_F * f2_ * t)
            + __expf(-PI_F * b3_ * t) * __sinf(2.f * PI_F * f3_ * t);
    comb[k] = v;
    mx = fmaxf(mx, fabsf(v));
  }
  red[tid] = mx; __syncthreads();
  for (int s = 128; s > 0; s >>= 1) { if (tid < s) red[tid] = fmaxf(red[tid], red[tid + s]); __syncthreads(); }
  float scale = red[0] + 1e-8f;
  for (int j = tid; j < KTAP; j += 256)
    filtrev[(size_t)row * KTAP + j] = comb[KTAP - 1 - j] / scale;
}

// ---------------- per-frame full convolution (exc ⊛ flipped filt) ----------------
// Staging via async global->LDS (all transfers exactly in-range).
__global__ void frameconv_kernel(const float* __restrict__ exc, const float* __restrict__ filtrev,
                                 float* __restrict__ conv) {
  __shared__ float xs[HOP];
  __shared__ float gs[KTAP];
  int row = blockIdx.x, tid = threadIdx.x;
#ifdef USE_ASYNC_LDS
  if (tid < 16)
    lds_async_b128(xs + tid * 4, exc + (size_t)row * HOP + tid * 4);
  for (int q = tid; q < KTAP / 4; q += 256)
    lds_async_b128(gs + q * 4, filtrev + (size_t)row * KTAP + q * 4);
  wait_async0();
  __syncthreads();
#else
  xs[tid] = exc[(size_t)row * HOP + tid];
  for (int j = tid; j < KTAP; j += 256) gs[j] = filtrev[(size_t)row * KTAP + j];
  __syncthreads();
#endif
  for (int t = tid; t < LCONV; t += 256) {
    int lo = t - (KTAP - 1); if (lo < 0) lo = 0;
    int hi = t < (HOP - 1) ? t : (HOP - 1);
    float acc = 0.f;
    for (int k = lo; k <= hi; k++) acc += xs[k] * gs[t - k];
    conv[(size_t)row * 1280 + t] = acc;
  }
}

// ---------------- overlap-add (gather, deterministic) ----------------
__global__ void ola_kernel(const float* __restrict__ conv, float* __restrict__ audio) {
  size_t idx = (size_t)blockIdx.x * blockDim.x + threadIdx.x;
  if (idx >= (size_t)NB * TL) return;
  int b = (int)(idx / TL);
  int n = (int)(idx % TL);
  int fhi = n / HOP; if (fhi > FRM - 1) fhi = FRM - 1;
  int flo = (n > LCONV - 1) ? ((n - (LCONV - 1) + HOP - 1) / HOP) : 0;
  float acc = 0.f;
  for (int f = flo; f <= fhi; f++)
    acc += conv[(size_t)(b * FRM + f) * 1280 + (n - f * HOP)];
  audio[(size_t)b * TLPAD + n] = acc;
}

// ---------------- reverb: direct 24000-tap conv + wet mix ----------------
__global__ void reverb_kernel(const float* __restrict__ audio, const float* __restrict__ ir,
                              const float* __restrict__ wet_logit, float* __restrict__ out) {
  __shared__ float irs[256];
  __shared__ float as[512];
  int b = blockIdx.y, tid = threadIdx.x;
  int n0 = blockIdx.x * 256;
  int n = n0 + tid;
  const float* a = audio + (size_t)b * TLPAD;
  float acc = 0.f;
  const int NCH = (REVN + 255) / 256;
  for (int c = 0; c < NCH; c++) {
    int m0 = c * 256;
    int mi = m0 + tid;
    irs[tid] = (mi < REVN) ? ir[mi] : 0.f;
    int base = n0 - m0 - 255;
    int j0 = base + tid, j1 = base + tid + 256;
    as[tid]       = (j0 >= 0 && j0 < TL) ? a[j0] : 0.f;
    as[tid + 256] = (j1 >= 0 && j1 < TL) ? a[j1] : 0.f;
    __syncthreads();
    // prefetch next chunk while this one computes (global_prefetch_b8)
    if (c + 1 < NCH) {
      int mn = m0 + 256 + tid;
      if (mn < REVN) __builtin_prefetch(ir + mn, 0, 1);
      int jn = base - 256 + tid;
      if (jn >= 0 && jn < TL) __builtin_prefetch(a + jn, 0, 1);
    }
#pragma unroll 8
    for (int i = 0; i < 256; i++) acc += irs[i] * as[tid + 255 - i];
    __syncthreads();
  }
  if (n < TL) {
    float wet = sigmoidf_(wet_logit[0]);
    out[(size_t)b * TL + n] = (1.f - wet) * a[n] + wet * acc;
  }
}

// ---------------- host launcher ----------------
extern "C" void kernel_launch(void* const* d_in, const int* in_sizes, int n_in,
                              void* d_out, int out_size, void* d_ws, size_t ws_size,
                              hipStream_t stream) {
  const float* f0    = (const float*)d_in[0];
  const float* loud  = (const float*)d_in[1];
  const float* gend  = (const float*)d_in[2];
  const float* age   = (const float*)d_in[3];
  const float* z     = (const float*)d_in[4];
  const float* noise = (const float*)d_in[5];
  const float* W_in  = (const float*)d_in[6];
  const float* b_in  = (const float*)d_in[7];
  const float* W_ih  = (const float*)d_in[8];
  const float* W_hh  = (const float*)d_in[9];
  const float* b_ih  = (const float*)d_in[10];
  const float* b_hh  = (const float*)d_in[11];
  const float* W_oq  = (const float*)d_in[12];
  const float* b_oq  = (const float*)d_in[13];
  const float* W_vg  = (const float*)d_in[14];
  const float* b_vg  = (const float*)d_in[15];
  const float* W_ug  = (const float*)d_in[16];
  const float* b_ug  = (const float*)d_in[17];
  const float* W_ff  = (const float*)d_in[18];
  const float* b_ff  = (const float*)d_in[19];
  const float* W_fb  = (const float*)d_in[20];
  const float* b_fb  = (const float*)d_in[21];
  const float* rir   = (const float*)d_in[22];
  const float* wetl  = (const float*)d_in[23];

  float* out = (float*)d_out;
  const size_t NROW = (size_t)NB * FRM; // 8192
  float* oq_o = out + (size_t)NB * TL;
  float* vg_o = oq_o + NROW;
  float* ug_o = vg_o + NROW;
  float* ff_o = ug_o + NROW;
  float* fb_o = ff_o + NROW * 4;

  // bump allocator over d_ws
  char* p = (char*)d_ws;
  auto alloc = [&](size_t bytes) -> void* {
    void* r = (void*)p;
    p += (bytes + 255) & ~(size_t)255;
    return r;
  };
  float* mean   = (float*)alloc(NB * sizeof(float));
  float* start  = (float*)alloc(NROW * sizeof(float));
  h16*   x28h   = (h16*)  alloc(NROW * 32 * sizeof(h16));
  h16*   Bp_in  = (h16*)  alloc((size_t)32 * 1 * 512 * sizeof(h16));
  h16*   Bp_ih  = (h16*)  alloc((size_t)96 * 16 * 512 * sizeof(h16));
  h16*   Bp_hh  = (h16*)  alloc((size_t)96 * 16 * 512 * sizeof(h16));
  h16*   xh     = (h16*)  alloc(NROW * RCH * sizeof(h16));
  float* GI     = (float*)alloc(NROW * 3 * RCH * sizeof(float));   // 50.3 MB
  float* Hall   = (float*)alloc(NROW * RCH * sizeof(float));
  float* exc    = (float*)alloc(NROW * HOP * sizeof(float));
  float* filtr  = (float*)alloc(NROW * KTAP * sizeof(float));
  float* audio  = (float*)alloc((size_t)NB * TLPAD * sizeof(float));
  float* conv   = GI; // GI consumed by GRU before frameconv runs: safe alias (41.9MB < 50.3MB)

  // 1) controller prep
  mean_f0_kernel<<<NB, 256, 0, stream>>>(f0, mean);
  scan_start_kernel<<<NB, 1024, 0, stream>>>(f0, start);
  build_x28_kernel<<<(int)(NROW / 256), 256, 0, stream>>>(f0, loud, gend, age, z, mean, x28h);

  // 2) pack WMMA B operands
  pack_b_kernel<<<(32 * 1 * 32 + 255) / 256, 256, 0, stream>>>(W_in, 28, 28, 32, 1, Bp_in);
  pack_b_kernel<<<(96 * 16 * 32 + 255) / 256, 256, 0, stream>>>(W_ih, 512, 512, 96, 16, Bp_ih);
  pack_b_kernel<<<(96 * 16 * 32 + 255) / 256, 256, 0, stream>>>(W_hh, 512, 512, 96, 16, Bp_hh);

  // 3) x = relu(in28 @ W_in^T + b_in)   [8192 x 512], f16 out for next GEMM
  wmma_gemm_kernel<<<512 * 32 / 8, 256, 0, stream>>>(x28h, 32, Bp_in, 1, b_in,
                                                     nullptr, xh, RCH, 32, 1);
  // 4) GI = x @ W_ih^T + b_ih           [8192 x 1536]
  wmma_gemm_kernel<<<512 * 96 / 8, 256, 0, stream>>>(xh, RCH, Bp_ih, 16, b_ih,
                                                     GI, nullptr, 3 * RCH, 96, 0);
  // 5) sequential GRU (single persistent WGP, WMMA recurrence + async GI staging)
  gru_kernel<<<1, 1024, 0, stream>>>(Bp_hh, GI, b_hh, Hall);

  // 6) output heads -> d_out regions
  heads_kernel<<<(int)NROW, 256, 0, stream>>>(Hall, W_oq, b_oq, W_vg, b_vg, W_ug, b_ug,
                                              W_ff, b_ff, W_fb, b_fb,
                                              oq_o, vg_o, ug_o, ff_o, fb_o);
  // 7) excitation
  excitation_kernel<<<(int)NROW, HOP, 0, stream>>>(f0, start, noise, oq_o, vg_o, ug_o, exc);
  // 8) time-varying FIR (normalized + pre-flipped)
  filter_kernel<<<(int)NROW, 256, 0, stream>>>(ff_o, fb_o, filtr);
  // 9) per-frame full convolution (async LDS staging)
  frameconv_kernel<<<(int)NROW, 256, 0, stream>>>(exc, filtr, conv);
  // 10) overlap-add (gather)
  ola_kernel<<<(int)(((size_t)NB * TL + 255) / 256), 256, 0, stream>>>(conv, audio);
  // 11) reverb + wet mix -> out
  dim3 rgrid((TL + 255) / 256, NB);
  reverb_kernel<<<rgrid, 256, 0, stream>>>(audio, rir, wetl, out);
}
